// ContextParallelAttention_82892868813318
// MI455X (gfx1250) — matrix operand order repaired and checked by reading, not compiled, not run on hardware
//
#include <hip/hip_runtime.h>
#include <hip/hip_bf16.h>
#include <math.h>

// ---------------------------------------------------------------------------
// Causal multi-head attention block for MI455X (gfx1250), bf16 WMMA pipeline.
//   y = ( softmax_causal( (xWq')(xWk')^T / sqrt(128) ) (xWv') ) Wo'
// Matmuls via v_wmma_f32_16x16x32_bf16; tile staging via
// global_load_async_to_lds_b128 (ASYNCcnt) with double buffering.
// ---------------------------------------------------------------------------

typedef __attribute__((ext_vector_type(16))) __bf16 v16bf;
typedef __attribute__((ext_vector_type(8)))  __bf16 v8bf;
typedef __attribute__((ext_vector_type(8)))  float  v8f;

#define D_MODEL 2048
#define NHEADS  16
#define HDIM    128
#define BATCH   2
#define SEQ     2048
#define MROWS   (BATCH * SEQ)   // 4096

// Generic shared pointer -> 32-bit LDS byte offset (flat LDS aperture keeps the
// offset in addr[31:0]).
__device__ __forceinline__ uint32_t lds_off(const void* p) {
  return (uint32_t)(uintptr_t)p;
}

// Async DMA: 16 bytes global -> LDS, tracked by ASYNCcnt (no VGPR data path).
__device__ __forceinline__ void async_copy_b128(uint32_t lds_byte_off,
                                                const void* gptr) {
  asm volatile("global_load_async_to_lds_b128 %0, %1, off"
               :: "v"(lds_byte_off), "v"(gptr)
               : "memory");
}

__device__ __forceinline__ void wait_async0() {
  asm volatile("s_wait_asynccnt 0x0" ::: "memory");
}

// Assemble a 16-bit A-operand fragment: first 8 elems at K-offset (lane>>4)*16,
// next 8 at (lane>>4)*16 + 8  (contiguous k = 16*(lane>>4) + s, same order as the B loads).
__device__ __forceinline__ v16bf make_frag(const __bf16* p_lo, const __bf16* p_hi) {
  v8bf lo = *(const v8bf*)p_lo;
  v8bf hi = *(const v8bf*)p_hi;
  v16bf r;
#pragma unroll
  for (int i = 0; i < 8; ++i) { r[i] = lo[i]; r[i + 8] = hi[i]; }
  return r;
}

// ---------------------------------------------------------------------------
__global__ void cvt_f32_bf16(const float* __restrict__ in,
                             __bf16* __restrict__ out, int n) {
  int i = blockIdx.x * blockDim.x + threadIdx.x;
  if (i < n) out[i] = (__bf16)in[i];
}

// ---------------------------------------------------------------------------
// C[M,N] = A[M,K] * W[N,K]^T   (both row-major, K contiguous)
// Block: 256 threads = 8 waves (4x2). C-tile 128x128, wave-tile 32x64
// (acc[2][4] => 8 WMMAs / K-step / wave). A tile double-buffered in LDS via
// async-to-LDS DMA; B fragments are contiguous 32B loads straight from W.
template <bool F32OUT>
__global__ __launch_bounds__(256)
void gemm_xwt(const __bf16* __restrict__ A, const __bf16* __restrict__ W,
              void* __restrict__ C, int M, int N, int K) {
  __shared__ __attribute__((aligned(16))) __bf16 Atile[2][128 * 32];

  const int t     = threadIdx.x;
  const int lane  = t & 31;
  const int wid   = t >> 5;
  const int waveM = wid >> 1;   // 0..3 -> 32 rows each
  const int waveN = wid & 1;    // 0..1 -> 64 cols each
  const int mbase = blockIdx.x * 128;
  const int nbase = blockIdx.y * 128;
  const int lrow  = lane & 15;
  const int lhalf = lane >> 4;

  // staging coords: 512 x b128 chunks per tile, 2 per thread
  const int srow0 = (t + 0)   >> 2, squad0 = (t + 0)   & 3;
  const int srow1 = (t + 256) >> 2, squad1 = (t + 256) & 3;
  const uint32_t l0 = lds_off(&Atile[0][srow0 * 32 + squad0 * 8]);
  const uint32_t l1 = lds_off(&Atile[0][srow1 * 32 + squad1 * 8]);
  const uint32_t bufstep = 128 * 32 * 2;  // bytes per buffer

  v8f acc[2][4] = {};

  // prologue: async-stage tile kb=0 into buffer 0
  async_copy_b128(l0, A + (size_t)(mbase + srow0) * K + squad0 * 8);
  async_copy_b128(l1, A + (size_t)(mbase + srow1) * K + squad1 * 8);

  int parity = 0;
  for (int kb = 0; kb < K; kb += 32) {
    wait_async0();        // own wave's DMA chunks have landed
    __syncthreads();      // whole tile visible to all waves

    if (kb + 32 < K) {    // overlap: DMA next tile into the other buffer
      uint32_t bo = (uint32_t)(parity ^ 1) * bufstep;
      async_copy_b128(l0 + bo, A + (size_t)(mbase + srow0) * K + (kb + 32) + squad0 * 8);
      async_copy_b128(l1 + bo, A + (size_t)(mbase + srow1) * K + (kb + 32) + squad1 * 8);
    }

    const __bf16* At = Atile[parity];
    v16bf afrag[2];
#pragma unroll
    for (int tm = 0; tm < 2; ++tm) {
      const __bf16* p = &At[(waveM * 32 + tm * 16 + lrow) * 32 + lhalf * 16];
      afrag[tm] = make_frag(p, p + 8);
    }
#pragma unroll
    for (int tn = 0; tn < 4; ++tn) {
      const int ncol = nbase + waveN * 64 + tn * 16 + lrow;
      const __bf16* bp = W + (size_t)ncol * K + kb + lhalf * 16;
      v16bf bfrag = *(const v16bf*)bp;   // 32B contiguous
#pragma unroll
      for (int tm = 0; tm < 2; ++tm)
        acc[tm][tn] = __builtin_amdgcn_wmma_f32_16x16x32_bf16(
            false, afrag[tm], false, bfrag, (short)0, acc[tm][tn], false, false);
    }
    parity ^= 1;
  }

  // ---- store: C VGPR r -> row r + 8*(lane>=16), col lane&15 ----
#pragma unroll
  for (int tm = 0; tm < 2; ++tm)
#pragma unroll
    for (int tn = 0; tn < 4; ++tn)
#pragma unroll
      for (int r = 0; r < 8; ++r) {
        int gr = mbase + waveM * 32 + tm * 16 + r + lhalf * 8;
        int gc = nbase + waveN * 64 + tn * 16 + lrow;
        float v = acc[tm][tn][r];
        if (F32OUT) ((float*)C)[(size_t)gr * N + gc] = v;
        else        ((__bf16*)C)[(size_t)gr * N + gc] = (__bf16)v;
      }
}

// ---------------------------------------------------------------------------
// Attention: one block per (q-tile of 16 rows, head, batch). 8 waves.
// Full score row block [16 x SEQ] f32 in dynamic LDS (128 KB; WGP has 320 KB),
// then masked softmax, P as bf16 in LDS, then O = P @ V via WMMA.
__global__ __launch_bounds__(256)
void attn_kernel(const __bf16* __restrict__ Q, const __bf16* __restrict__ Kb,
                 const __bf16* __restrict__ V, __bf16* __restrict__ O) {
  extern __shared__ __attribute__((aligned(16))) char smem[];
  float*  sS = (float*)smem;                                  // 16*SEQ f32
  __bf16* sP = (__bf16*)(smem + 16 * SEQ * 4);                // 16*SEQ bf16
  __bf16* sQ = (__bf16*)(smem + 16 * SEQ * 4 + 16 * SEQ * 2); // 16*HDIM bf16

  const int t = threadIdx.x, lane = t & 31, wid = t >> 5;
  const int lrow = lane & 15, lhalf = lane >> 4;
  const int qt = blockIdx.x, h = blockIdx.y, b = blockIdx.z;
  const int qbase = qt * 16;
  const size_t seqbase = (size_t)b * SEQ;
  const float scale = 0.08838834764831845f;  // 1/sqrt(128)

  // ---- async-DMA Q tile 16 x 128 (bf16) into LDS ----
  {
    int row = t >> 4, chunk = t & 15;
    async_copy_b128(lds_off(&sQ[row * HDIM + chunk * 8]),
                    Q + (seqbase + qbase + row) * D_MODEL + h * HDIM + chunk * 8);
    wait_async0();
  }
  __syncthreads();

  // ---- scores: wave w owns key tiles [w*16, w*16+16); causal skip is
  //      wave-uniform so EXEC stays all-1s for WMMA ----
  const int qhi = qbase + 15;
#pragma unroll 1
  for (int ti = 0; ti < 16; ++ti) {
    int kt = wid * 16 + ti;
    int kcol0 = kt * 16;
    if (kcol0 > qhi) break;  // remaining tiles fully masked
    v8f c = {};
#pragma unroll
    for (int kk = 0; kk < 4; ++kk) {       // Dh = 128 = 4 x 32
      const __bf16* qp = &sQ[lrow * HDIM + kk * 32 + lhalf * 16];
      v16bf a = make_frag(qp, qp + 8);
      // B operand = K^T column = contiguous row of K in memory
      const __bf16* kp = Kb + (seqbase + kcol0 + lrow) * D_MODEL +
                         h * HDIM + kk * 32 + lhalf * 16;
      v16bf bb = *(const v16bf*)kp;
      c = __builtin_amdgcn_wmma_f32_16x16x32_bf16(false, a, false, bb,
                                                  (short)0, c, false, false);
    }
#pragma unroll
    for (int r = 0; r < 8; ++r)
      sS[(r + lhalf * 8) * SEQ + kcol0 + lrow] = c[r] * scale;
  }
  __syncthreads();

  // ---- masked softmax: 16 threads per row, shfl_xor reductions ----
  {
    int row = t >> 4, lic = t & 15;
    int q_abs = qbase + row;
    float* srow = sS + row * SEQ;
    __bf16* prow = sP + row * SEQ;
    float m = -INFINITY;
    for (int c = lic; c <= q_abs; c += 16) m = fmaxf(m, srow[c]);
#pragma unroll
    for (int off = 1; off < 16; off <<= 1) m = fmaxf(m, __shfl_xor(m, off, 32));
    float l = 0.0f;
    for (int c = lic; c <= q_abs; c += 16) l += expf(srow[c] - m);
#pragma unroll
    for (int off = 1; off < 16; off <<= 1) l += __shfl_xor(l, off, 32);
    float inv = 1.0f / l;
    for (int c = lic; c < SEQ; c += 16) {
      if (c <= q_abs) prow[c] = (__bf16)(expf(srow[c] - m) * inv);
      else            prow[c] = (__bf16)0.0f;
    }
  }
  __syncthreads();

  // ---- O = P @ V : wave w -> head-dim columns [w*16, w*16+16) ----
  {
    const int ncol = h * HDIM + wid * 16 + lrow;
    const __bf16* vbase = V + seqbase * D_MODEL + ncol;
    v8f c = {};
    const int klimit = ((qbase + 16 + 31) >> 5) << 5;  // causal K-loop bound
    for (int kb = 0; kb < klimit; kb += 32) {
      const __bf16* pp = &sP[lrow * SEQ + kb + lhalf * 16];
      v16bf a = make_frag(pp, pp + 8);
      v16bf bb;
#pragma unroll
      for (int j = 0; j < 16; ++j)  // column of V: strided gather
        bb[j] = vbase[(size_t)(kb + lhalf * 16 + j) * D_MODEL];
      c = __builtin_amdgcn_wmma_f32_16x16x32_bf16(false, a, false, bb,
                                                  (short)0, c, false, false);
    }
#pragma unroll
    for (int r = 0; r < 8; ++r)
      O[(seqbase + qbase + r + lhalf * 8) * D_MODEL + ncol] = (__bf16)c[r];
  }
}

// ---------------------------------------------------------------------------
extern "C" void kernel_launch(void* const* d_in, const int* in_sizes, int n_in,
                              void* d_out, int out_size, void* d_ws, size_t ws_size,
                              hipStream_t stream) {
  const float* x  = (const float*)d_in[0];
  const float* wq = (const float*)d_in[1];
  const float* wk = (const float*)d_in[2];
  const float* wv = (const float*)d_in[3];
  const float* wo = (const float*)d_in[4];

  const size_t NX = (size_t)MROWS * D_MODEL;    // 8,388,608
  const size_t NW = (size_t)D_MODEL * D_MODEL;  // 4,194,304

  char* ws = (char*)d_ws;
  __bf16* xb  = (__bf16*)ws; ws += NX * 2;
  __bf16* wqb = (__bf16*)ws; ws += NW * 2;
  __bf16* wkb = (__bf16*)ws; ws += NW * 2;
  __bf16* wvb = (__bf16*)ws; ws += NW * 2;
  __bf16* wob = (__bf16*)ws; ws += NW * 2;
  __bf16* qb  = (__bf16*)ws; ws += NX * 2;
  __bf16* kb  = (__bf16*)ws; ws += NX * 2;
  __bf16* vb  = (__bf16*)ws; ws += NX * 2;
  __bf16* ab  = (__bf16*)ws; ws += NX * 2;   // attention output (pre-Wo)

  // fp32 -> bf16 downconversion
  cvt_f32_bf16<<<(unsigned)((NX + 255) / 256), 256, 0, stream>>>(x,  xb,  (int)NX);
  cvt_f32_bf16<<<(unsigned)((NW + 255) / 256), 256, 0, stream>>>(wq, wqb, (int)NW);
  cvt_f32_bf16<<<(unsigned)((NW + 255) / 256), 256, 0, stream>>>(wk, wkb, (int)NW);
  cvt_f32_bf16<<<(unsigned)((NW + 255) / 256), 256, 0, stream>>>(wv, wvb, (int)NW);
  cvt_f32_bf16<<<(unsigned)((NW + 255) / 256), 256, 0, stream>>>(wo, wob, (int)NW);

  // Q/K/V projections
  dim3 gg(MROWS / 128, D_MODEL / 128);
  gemm_xwt<false><<<gg, 256, 0, stream>>>(xb, wqb, qb, MROWS, D_MODEL, D_MODEL);
  gemm_xwt<false><<<gg, 256, 0, stream>>>(xb, wkb, kb, MROWS, D_MODEL, D_MODEL);
  gemm_xwt<false><<<gg, 256, 0, stream>>>(xb, wvb, vb, MROWS, D_MODEL, D_MODEL);

  // causal attention
  size_t smem = (size_t)16 * SEQ * 4 + (size_t)16 * SEQ * 2 + (size_t)16 * HDIM * 2;
  attn_kernel<<<dim3(SEQ / 16, NHEADS, BATCH), 256, smem, stream>>>(qb, kb, vb, ab);

  // output projection (f32 out)
  gemm_xwt<true><<<gg, 256, 0, stream>>>(ab, wob, d_out, MROWS, D_MODEL, D_MODEL);
}
